// GATv2ActorCritic_76390288326780
// MI455X (gfx1250) — compile-verified, matrix operand order checked by reading, not run on hardware
//
#include <hip/hip_runtime.h>
#include <hip/hip_fp16.h>

typedef __attribute__((ext_vector_type(16))) _Float16 v16h;
typedef __attribute__((ext_vector_type(8)))  _Float16 v8h;
typedef __attribute__((ext_vector_type(8)))  float    v8f;

// ---------- helpers: monotone float<->uint mapping for atomic max ----------
__device__ __forceinline__ unsigned fp_ord(float f) {
    unsigned u = __float_as_uint(f);
    return (u & 0x80000000u) ? ~u : (u | 0x80000000u);
}
__device__ __forceinline__ float fp_unord(unsigned u) {
    unsigned b = (u & 0x80000000u) ? (u ^ 0x80000000u) : ~u;
    return __uint_as_float(b);
}

// ---------- layer-1 node transforms: x(N,2) -> xl1,xr1 (N,128) ----------
__global__ void k_node_transform1(const float* __restrict__ x,
                                  const float* __restrict__ Wl, const float* __restrict__ bl,
                                  const float* __restrict__ Wr, const float* __restrict__ br,
                                  float* __restrict__ xl1, float* __restrict__ xr1, int N) {
    int idx = blockIdx.x * blockDim.x + threadIdx.x;
    if (idx >= N * 128) return;
    int n = idx >> 7, d = idx & 127;
    float x0 = x[n * 2], x1 = x[n * 2 + 1];
    xl1[idx] = fmaf(x0, Wl[d], fmaf(x1, Wl[128 + d], bl[d]));
    xr1[idx] = fmaf(x0, Wr[d], fmaf(x1, Wr[128 + d], br[d]));
}

// ---------- edge attention logits + segment max (mapped-uint atomicMax) ----------
template <int H, int C>
__global__ void k_edge_logits(const float* __restrict__ xl, const float* __restrict__ xr,
                              const float* __restrict__ ea, const float* __restrict__ We,
                              const float* __restrict__ att,
                              const int* __restrict__ src, const int* __restrict__ tgt,
                              float* __restrict__ elog, unsigned* __restrict__ emaxM, int E) {
    const int D = H * C;
    int idx = blockIdx.x * blockDim.x + threadIdx.x;
    if (idx >= E * H) return;
    int e = idx / H, h = idx % H;
    int s = src[e], t = tgt[e];
    float a0 = ea[e * 2], a1 = ea[e * 2 + 1];
    int base = h * C;
    float acc = 0.f;
#pragma unroll
    for (int c = 0; c < C; ++c) {
        int d = base + c;
        float m = xl[s * D + d] + xr[t * D + d] + a0 * We[d] + a1 * We[D + d];
        m = (m > 0.f) ? m : 0.2f * m;          // leaky_relu(0.2)
        acc = fmaf(m, att[base + c], acc);
    }
    elog[idx] = acc;
    atomicMax(&emaxM[t * H + h], fp_ord(acc));
}

// ---------- exp(e - max) + segment sum of denominators ----------
template <int H>
__global__ void k_edge_exp(float* __restrict__ elog, const unsigned* __restrict__ emaxM,
                           float* __restrict__ denom, const int* __restrict__ tgt, int E) {
    int idx = blockIdx.x * blockDim.x + threadIdx.x;
    if (idx >= E * H) return;
    int e = idx / H, h = idx % H;
    int t = tgt[e];
    float mx = fp_unord(emaxM[t * H + h]);
    float v = __expf(elog[idx] - mx);
    elog[idx] = v;                              // in-place: logits -> exp weights
    atomicAdd(&denom[t * H + h], v);
}

// ---------- weighted scatter-add aggregation (4 channels per thread) ----------
template <int H, int C>
__global__ void k_edge_agg(const float* __restrict__ ee, const float* __restrict__ denom,
                           const float* __restrict__ xl,
                           const int* __restrict__ src, const int* __restrict__ tgt,
                           float* __restrict__ agg, int E) {
    const int D = H * C;
    const int Q = D / 4;
    int idx = blockIdx.x * blockDim.x + threadIdx.x;
    if (idx >= E * Q) return;
    int e = idx / Q, q = idx % Q;
    int d = q * 4, h = d / C;
    int s = src[e], t = tgt[e];
    float alpha = ee[e * H + h] / denom[t * H + h];
    float4 v = *(const float4*)(xl + s * D + d);
    atomicAdd(&agg[t * D + d + 0], alpha * v.x);
    atomicAdd(&agg[t * D + d + 1], alpha * v.y);
    atomicAdd(&agg[t * D + d + 2], alpha * v.z);
    atomicAdd(&agg[t * D + d + 3], alpha * v.w);
}

// ---------- ELU(agg + bias1) and convert to f16 for the WMMA GEMM ----------
__global__ void k_elu_half(const float* __restrict__ agg, const float* __restrict__ bias,
                           _Float16* __restrict__ hh, int N) {
    int idx = blockIdx.x * blockDim.x + threadIdx.x;
    if (idx >= N * 128) return;
    int d = idx & 127;
    float v = agg[idx] + bias[d];
    v = (v > 0.f) ? v : (__expf(v) - 1.f);      // elu
    hh[idx] = (_Float16)v;
}

// ---------- pack W_l2 / W_r2 (128x64 f32) into per-lane WMMA B fragments ----------
// Layout: Wpk[which * 8192 + ((kc*4 + nt)*32 + lane)*16 + j]  (v16h per lane, 32B)
// j<8  -> K = kc*32 + half_sel*8      + j   (lanes 0-15: K 0-7,  lanes 16-31: K 8-15)
// j>=8 -> K = kc*32 + 16 + half_sel*8 + j-8 (lanes 0-15: K 16-23, lanes 16-31: K 24-31)
__global__ void k_w_pack(const float* __restrict__ Wl, const float* __restrict__ Wr,
                         _Float16* __restrict__ Wpk) {
    int idx = blockIdx.x * blockDim.x + threadIdx.x;
    if (idx >= 2 * 8192) return;
    int which = idx >> 13;
    int r = idx & 8191;
    int j = r & 15;
    int lane = (r >> 4) & 31;
    int frag = r >> 9;                   // 0..15 = kc*4 + nt
    int kc = frag >> 2, nt = frag & 3;
    int half_sel = lane >> 4, lidx = lane & 15;
    int kb = (j < 8) ? (half_sel * 8) : (16 + half_sel * 8);
    int k = kc * 32 + kb + (j & 7);
    int n = nt * 16 + lidx;
    const float* W = which ? Wr : Wl;
    Wpk[idx] = (_Float16)W[k * 64 + n];
}

// ---------- layer-2 dense transforms via v_wmma_f32_16x16x32_f16 ----------
// One wave per 16-row tile of h (N x 128). K = 128 => 4 k-chunks of 32; Nout = 64
// => 4 n-tiles of 16; two weight matrices => 32 WMMA per wave. B fragments are
// staged in LDS in fragment order (one 32B LDS read per fragment).
__global__ void k_gemm2_wmma(const _Float16* __restrict__ hh,
                             const _Float16* __restrict__ Wpk,
                             const float* __restrict__ bl, const float* __restrict__ br,
                             float* __restrict__ xl2, float* __restrict__ xr2, int N) {
    __shared__ __align__(32) _Float16 ldsW[2 * 8192];   // 32 KB: packed Wl then Wr

    // cooperative copy of packed weights into LDS (2048 x 16B)
    {
        const uint4* g = (const uint4*)Wpk;
        uint4* l = (uint4*)ldsW;
        for (int i = threadIdx.x; i < 2048; i += blockDim.x) l[i] = g[i];
    }
    __syncthreads();

    int wave = (blockIdx.x * blockDim.x + threadIdx.x) >> 5;
    int lane = threadIdx.x & 31;
    int row0 = wave * 16;
    if (row0 >= N) return;                      // wave-uniform exit

    int half_sel = lane >> 4;                   // 0: lanes 0-15, 1: lanes 16-31
    int lidx = lane & 15;
    int kb_lo = half_sel * 8;                   // K 0-7 vs 8-15 in first half
    int kb_hi = 16 + kb_lo;                     // K 16-23 vs 24-31 in second half
    int m = row0 + lidx;
    if (m >= N) m = N - 1;                      // safe clamp (stores guarded below)

    const v16h* fragL = (const v16h*)ldsW;         // 16 fragments x 32 lanes
    const v16h* fragR = (const v16h*)(ldsW + 8192);

    v8f accl[4] = {};                           // 4 n-tiles, each 16x16 f32 accum
    v8f accr[4] = {};

#pragma unroll
    for (int kc = 0; kc < 4; ++kc) {
        const _Float16* pa = hh + m * 128 + kc * 32;
        v8h alo = *(const v8h*)(pa + kb_lo);
        v8h ahi = *(const v8h*)(pa + kb_hi);
        v16h a = __builtin_shufflevector(alo, ahi, 0, 1, 2, 3, 4, 5, 6, 7,
                                         8, 9, 10, 11, 12, 13, 14, 15);
#pragma unroll
        for (int nt = 0; nt < 4; ++nt) {
            int fi = (kc * 4 + nt) * 32 + lane;
            v16h b  = fragL[fi];
            v16h b2 = fragR[fi];
            accl[nt] = __builtin_amdgcn_wmma_f32_16x16x32_f16(
                false, a, false, b, (short)0, accl[nt], false, false);
            accr[nt] = __builtin_amdgcn_wmma_f32_16x16x32_f16(
                false, a, false, b2, (short)0, accr[nt], false, false);
        }
    }
#pragma unroll
    for (int nt = 0; nt < 4; ++nt) {
        int n = nt * 16 + lidx;
        float bln = bl[n], brn = br[n];
#pragma unroll
        for (int r = 0; r < 8; ++r) {
            int mo = row0 + r + half_sel * 8;   // C/D layout: VGPR r -> row r (+8 for hi lanes)
            if (mo < N) {
                xl2[mo * 64 + n] = accl[nt][r] + bln;
                xr2[mo * 64 + n] = accr[nt][r] + brn;
            }
        }
    }
}

// ---------- global mean pool ----------
__global__ void k_pool(const float* __restrict__ agg2, const float* __restrict__ bias2,
                       const int* __restrict__ batch,
                       float* __restrict__ sums, float* __restrict__ cnts, int N) {
    int idx = blockIdx.x * blockDim.x + threadIdx.x;
    if (idx >= N * 64) return;
    int n = idx >> 6, d = idx & 63;
    int g = batch[n];
    atomicAdd(&sums[g * 64 + d], agg2[idx] + bias2[d]);
    if (d == 0) atomicAdd(&cnts[g], 1.0f);
}

__global__ void k_final(const float* __restrict__ sums, const float* __restrict__ cnts,
                        float* __restrict__ out, int G) {
    int idx = blockIdx.x * blockDim.x + threadIdx.x;
    if (idx >= G * 64) return;
    out[idx] = sums[idx] / fmaxf(cnts[idx >> 6], 1.0f);
}

// =============================== launcher ===============================
extern "C" void kernel_launch(void* const* d_in, const int* in_sizes, int n_in,
                              void* d_out, int out_size, void* d_ws, size_t ws_size,
                              hipStream_t stream) {
    const float* x      = (const float*)d_in[0];
    const float* ea     = (const float*)d_in[1];
    const float* W_l1   = (const float*)d_in[2];
    const float* b_l1   = (const float*)d_in[3];
    const float* W_r1   = (const float*)d_in[4];
    const float* b_r1   = (const float*)d_in[5];
    const float* W_e1   = (const float*)d_in[6];
    const float* att1   = (const float*)d_in[7];
    const float* bias1  = (const float*)d_in[8];
    const float* W_l2   = (const float*)d_in[9];
    const float* b_l2   = (const float*)d_in[10];
    const float* W_r2   = (const float*)d_in[11];
    const float* b_r2   = (const float*)d_in[12];
    const float* W_e2   = (const float*)d_in[13];
    const float* att2   = (const float*)d_in[14];
    const float* bias2  = (const float*)d_in[15];
    const int*   eidx   = (const int*)d_in[16];   // [2,E]: row0=src, row1=tgt
    const int*   batch  = (const int*)d_in[17];
    float* out = (float*)d_out;

    const int N = in_sizes[0] / 2;    // 50000
    const int E = in_sizes[1] / 2;    // 800000
    const int G = out_size / 64;      // 64
    const int* src = eidx;
    const int* tgt = eidx + E;

    // -------- workspace layout (accumulators first, zeroed in one memset) --------
    char* ws = (char*)d_ws;
    size_t off = 0;
    auto alloc = [&](size_t bytes) { size_t o = off; off = (off + bytes + 255) & ~(size_t)255; return o; };

    size_t o_emax1 = alloc((size_t)N * 16 * 4);
    size_t o_den1  = alloc((size_t)N * 16 * 4);
    size_t o_agg1  = alloc((size_t)N * 128 * 4);
    size_t o_emax2 = alloc((size_t)N * 4 * 4);
    size_t o_den2  = alloc((size_t)N * 4 * 4);
    size_t o_agg2  = alloc((size_t)N * 64 * 4);
    size_t o_sums  = alloc((size_t)G * 64 * 4);
    size_t o_cnts  = alloc((size_t)G * 4);
    size_t acc_bytes = off;

    size_t o_xl1 = alloc((size_t)N * 128 * 4);
    size_t o_xr1 = alloc((size_t)N * 128 * 4);
    size_t o_e1  = alloc((size_t)E * 16 * 4);
    size_t o_hh  = alloc((size_t)N * 128 * 2);
    size_t o_wpk = alloc((size_t)2 * 8192 * 2);   // packed f16 B fragments (Wl2 + Wr2)
    size_t o_xl2 = alloc((size_t)N * 64 * 4);
    size_t o_xr2 = alloc((size_t)N * 64 * 4);
    size_t o_e2  = alloc((size_t)E * 4 * 4);
    (void)ws_size;

    unsigned* emax1 = (unsigned*)(ws + o_emax1);
    float*    den1  = (float*)(ws + o_den1);
    float*    agg1  = (float*)(ws + o_agg1);
    unsigned* emax2 = (unsigned*)(ws + o_emax2);
    float*    den2  = (float*)(ws + o_den2);
    float*    agg2  = (float*)(ws + o_agg2);
    float*    sums  = (float*)(ws + o_sums);
    float*    cnts  = (float*)(ws + o_cnts);
    float*    xl1   = (float*)(ws + o_xl1);
    float*    xr1   = (float*)(ws + o_xr1);
    float*    e1    = (float*)(ws + o_e1);
    _Float16* hh    = (_Float16*)(ws + o_hh);
    _Float16* wpk   = (_Float16*)(ws + o_wpk);
    float*    xl2   = (float*)(ws + o_xl2);
    float*    xr2   = (float*)(ws + o_xr2);
    float*    e2    = (float*)(ws + o_e2);

    const int TB = 256;
    auto blocks = [&](long long t) { return (unsigned)((t + TB - 1) / TB); };

    // 0) zero all accumulators (mapped-uint 0 == -inf side for segment max)
    hipMemsetAsync(d_ws, 0, acc_bytes, stream);

    // ---- layer 1 ----
    k_node_transform1<<<blocks((long long)N * 128), TB, 0, stream>>>(
        x, W_l1, b_l1, W_r1, b_r1, xl1, xr1, N);
    k_edge_logits<16, 8><<<blocks((long long)E * 16), TB, 0, stream>>>(
        xl1, xr1, ea, W_e1, att1, src, tgt, e1, emax1, E);
    k_edge_exp<16><<<blocks((long long)E * 16), TB, 0, stream>>>(
        e1, emax1, den1, tgt, E);
    k_edge_agg<16, 8><<<blocks((long long)E * 32), TB, 0, stream>>>(
        e1, den1, xl1, src, tgt, agg1, E);
    k_elu_half<<<blocks((long long)N * 128), TB, 0, stream>>>(agg1, bias1, hh, N);

    // ---- layer-2 dense transforms (WMMA) ----
    k_w_pack<<<blocks(2 * 8192), TB, 0, stream>>>(W_l2, W_r2, wpk);
    long long gemm_threads = (long long)((N + 15) / 16) * 32;
    k_gemm2_wmma<<<blocks(gemm_threads), TB, 0, stream>>>(
        hh, wpk, b_l2, b_r2, xl2, xr2, N);

    // ---- layer 2 message passing ----
    k_edge_logits<4, 16><<<blocks((long long)E * 4), TB, 0, stream>>>(
        xl2, xr2, ea, W_e2, att2, src, tgt, e2, emax2, E);
    k_edge_exp<4><<<blocks((long long)E * 4), TB, 0, stream>>>(
        e2, emax2, den2, tgt, E);
    k_edge_agg<4, 16><<<blocks((long long)E * 16), TB, 0, stream>>>(
        e2, den2, xl2, src, tgt, agg2, E);

    // ---- global mean pool ----
    k_pool<<<blocks((long long)N * 64), TB, 0, stream>>>(agg2, bias2, batch, sums, cnts, N);
    k_final<<<blocks((long long)G * 64), TB, 0, stream>>>(sums, cnts, out, G);
}
